// ImplicitMLPAttention_68624987456109
// MI455X (gfx1250) — matrix operand order repaired and checked by reading, not compiled, not run on hardware
//
#include <hip/hip_runtime.h>
#include <hip/hip_bf16.h>

// ---------------------------------------------------------------------------
// Problem constants (B=4, N=1024, D=1024, H=16, DH=64, 3 layers)
// ---------------------------------------------------------------------------
#define BB   4
#define NN   1024
#define DD   1024
#define HH   16
#define DHH  64
#define HD   1024          // H*DH
#define NT   4096          // B*N
#define NLAYERS 3
#define RMS_EPS 1.1920929e-07f

typedef __attribute__((ext_vector_type(16))) __bf16 v16bf;
typedef __attribute__((ext_vector_type(8)))  float  v8f;

union F16x16 {              // one 16x32-bf16 A-frag / 32x16-bf16 B-frag per lane
    v16bf v;
    uint4 q[2];
    unsigned short u[16];
};

#define WMMA_BF16(a, b, c) \
    __builtin_amdgcn_wmma_f32_16x16x32_bf16(false, (a), false, (b), (short)0, (c), false, false)

// Async global->LDS copy path (CDNA5); falls back to direct loads if the
// builtin is not declared by this toolchain.
#if __has_builtin(__builtin_amdgcn_global_load_async_to_lds_b128) && \
    __has_builtin(__builtin_amdgcn_s_wait_asynccnt)
#define GEMM_ASYNC 1
typedef int v4i_vec __attribute__((__vector_size__(16)));
typedef __attribute__((address_space(1))) v4i_vec* gas_ptr_t;
typedef __attribute__((address_space(3))) v4i_vec* las_ptr_t;
#else
#define GEMM_ASYNC 0
#endif

static __device__ __forceinline__ unsigned short f2bf(float f) {
    unsigned int u = __float_as_uint(f);
    u = (u + 0x7FFFu + ((u >> 16) & 1u)) >> 16;      // round-to-nearest-even
    return (unsigned short)u;
}

// ---------------------------------------------------------------------------
// RMSNorm:   x = tokens * rsqrt(mean(tokens^2) + eps) * gamma   -> bf16
// ---------------------------------------------------------------------------
__global__ __launch_bounds__(256) void rmsnorm_kernel(
    const float* __restrict__ x, const float* __restrict__ gamma,
    unsigned short* __restrict__ out)
{
    __shared__ float red[256];
    const int row = blockIdx.x;
    const float* xr = x + (size_t)row * DD;
    float v[4]; float ss = 0.f;
#pragma unroll
    for (int j = 0; j < 4; ++j) { v[j] = xr[threadIdx.x + j * 256]; ss += v[j] * v[j]; }
    red[threadIdx.x] = ss;
    __syncthreads();
    for (int s = 128; s > 0; s >>= 1) {
        if (threadIdx.x < s) red[threadIdx.x] += red[threadIdx.x + s];
        __syncthreads();
    }
    const float scale = rsqrtf(red[0] * (1.0f / DD) + RMS_EPS);
    unsigned short* orow = out + (size_t)row * DD;
#pragma unroll
    for (int j = 0; j < 4; ++j) {
        const int c = threadIdx.x + j * 256;
        orow[c] = f2bf(v[j] * scale * gamma[c]);
    }
}

// ---------------------------------------------------------------------------
// one-shot weight transpose: fp32 [K,Nc] -> bf16 [Nc,K]   (LDS-tiled)
// ---------------------------------------------------------------------------
__global__ __launch_bounds__(256) void transpose_bf_kernel(
    const float* __restrict__ in, unsigned short* __restrict__ out, int K, int Nc)
{
    __shared__ unsigned short tile[32][33];
    const int tx = threadIdx.x & 31;
    const int ty = threadIdx.x >> 5;            // 0..7
    const int kb = blockIdx.y * 32, nb = blockIdx.x * 32;
#pragma unroll
    for (int j = 0; j < 32; j += 8)
        tile[ty + j][tx] = f2bf(in[(size_t)(kb + ty + j) * Nc + nb + tx]);
    __syncthreads();
#pragma unroll
    for (int j = 0; j < 32; j += 8)
        out[(size_t)(nb + ty + j) * K + kb + tx] = tile[tx][ty + j];
}

// ---------------------------------------------------------------------------
// SiLU in-place
// ---------------------------------------------------------------------------
__global__ void silu_kernel(float* __restrict__ x, int n)
{
    const int i = blockIdx.x * blockDim.x + threadIdx.x;
    if (i < n) { const float v = x[i]; x[i] = v / (1.0f + __expf(-v)); }
}

// ---------------------------------------------------------------------------
// RoPE (interleaved pairs, theta=10000) + fp32->bf16, layout [B,H,N,DH]
// ---------------------------------------------------------------------------
__global__ void rope_conv_kernel(const float* __restrict__ in,
                                 unsigned short* __restrict__ out, int total)
{
    const int idx = blockIdx.x * blockDim.x + threadIdx.x;
    if (idx >= total) return;
    const int pair = idx & 31;                 // DH/2 = 32 pairs
    const int pos  = (idx >> 5) & (NN - 1);
    const size_t off = (size_t)idx * 2;
    const float x0 = in[off], x1 = in[off + 1];
    const float ang = (float)pos * __expf(-(float)(2 * pair) * (9.210340372f / 64.0f));
    const float c = cosf(ang), s = sinf(ang);
    out[off]     = f2bf(x0 * c - x1 * s);
    out[off + 1] = f2bf(x1 * c + x0 * s);
}

// ---------------------------------------------------------------------------
// V conversion with transpose: fp32 [B,H,N,DH] -> bf16 [B,H,DH,N]
// (so PV B-fragments are per-lane key-contiguous straight from global)
// ---------------------------------------------------------------------------
__global__ void conv_vT_kernel(const float* __restrict__ in,
                               unsigned short* __restrict__ out, int total)
{
    const int i = blockIdx.x * blockDim.x + threadIdx.x;
    if (i >= total) return;
    const int dh  = i & (DHH - 1);
    const int pos = (i >> 6) & (NN - 1);
    const int bh  = i >> 16;
    out[((size_t)bh * DHH + dh) * NN + pos] = f2bf(in[i]);
}

// ---------------------------------------------------------------------------
// merge heads: [B,H,N,DH] fp32 -> [B,N,H*DH] bf16
// ---------------------------------------------------------------------------
__global__ void merge_kernel(const float* __restrict__ in,
                             unsigned short* __restrict__ out, int n)
{
    const int i = blockIdx.x * blockDim.x + threadIdx.x;
    if (i >= n) return;
    const int dh  = i & (DHH - 1);
    const int pos = (i >> 6) & (NN - 1);
    const int h   = (i >> 16) & (HH - 1);
    const int b   = i >> 20;
    out[(((size_t)b * NN + pos) * HD) + h * DHH + dh] = f2bf(in[i]);
}

// ---------------------------------------------------------------------------
// bf16 GEMM, f32 accumulate:  C[M,Nc] = A[M,K] @ Bt^T   (Bt is [Nc,K] bf16)
// block tile 128x64, 4 waves, each wave 32(M) x 64(N), K stepped by 32.
// A tile staged via async global->LDS (double buffered) when available.
// headSplit: write C as [B,H,N,DH] (row = b*N+n, col = h*64+dh)
// ---------------------------------------------------------------------------
__global__ __launch_bounds__(128) void gemm_bf16_kernel(
    const unsigned short* __restrict__ A, const unsigned short* __restrict__ Bt,
    float* __restrict__ C, int M, int K, int Nc, int headSplit)
{
    const int tid  = threadIdx.x;
    const int wave = tid >> 5, lane = tid & 31;
    const int lh = lane & 15, hi = lane >> 4;
    const int mbase = blockIdx.y * 128, nbase = blockIdx.x * 64;
    const int mw = mbase + wave * 32;
    const int ka0 = hi ? 8 : 0;           // A-frag K split per ISA layout table
    const int kb0 = hi ? 8 : 0;           // B-frag K split
    v8f acc[2][4] = {};

    const unsigned short* brow[4];
#pragma unroll
    for (int t = 0; t < 4; ++t)
        brow[t] = Bt + (size_t)(nbase + t * 16 + lh) * K + kb0;

#if GEMM_ASYNC
    __shared__ __align__(16) unsigned short Atile[2][128 * 32];
    auto stage = [&](int buf, int kb) {
#pragma unroll
        for (int j = 0; j < 4; ++j) {
            const int c   = j * 128 + tid;         // 512 chunks of 16B
            const int row = c >> 2, ko = (c & 3) * 8;
            const unsigned short* g = A + (size_t)(mbase + row) * K + kb + ko;
            __builtin_amdgcn_global_load_async_to_lds_b128(
                (gas_ptr_t)g, (las_ptr_t)&Atile[buf][row * 32 + ko], 0, 0);
        }
    };
    stage(0, 0);
    int buf = 0;
#else
    const unsigned short* arow0 = A + (size_t)(mw + lh) * K;
    const unsigned short* arow1 = A + (size_t)(mw + 16 + lh) * K;
#endif

    for (int kb = 0; kb < K; kb += 32) {
        F16x16 af0, af1;
#if GEMM_ASYNC
        if (kb + 32 < K) {
            stage(buf ^ 1, kb + 32);
            __builtin_amdgcn_s_wait_asynccnt(4);   // previous tile done, next in flight
        } else {
            __builtin_amdgcn_s_wait_asynccnt(0);
        }
        __syncthreads();
        const unsigned short* a0 = &Atile[buf][(wave * 32 + lh) * 32 + ka0];
        const unsigned short* a1 = &Atile[buf][(wave * 32 + 16 + lh) * 32 + ka0];
        af0.q[0] = *(const uint4*)(a0);  af0.q[1] = *(const uint4*)(a0 + 16);
        af1.q[0] = *(const uint4*)(a1);  af1.q[1] = *(const uint4*)(a1 + 16);
#else
        af0.q[0] = *(const uint4*)(arow0 + kb + ka0);
        af0.q[1] = *(const uint4*)(arow0 + kb + ka0 + 16);
        af1.q[0] = *(const uint4*)(arow1 + kb + ka0);
        af1.q[1] = *(const uint4*)(arow1 + kb + ka0 + 16);
#endif
#pragma unroll
        for (int t = 0; t < 4; ++t) {
            F16x16 bf_;
            bf_.q[0] = *(const uint4*)(brow[t] + kb);
            bf_.q[1] = *(const uint4*)(brow[t] + kb + 16);
            acc[0][t] = WMMA_BF16(af0.v, bf_.v, acc[0][t]);
            acc[1][t] = WMMA_BF16(af1.v, bf_.v, acc[1][t]);
        }
#if GEMM_ASYNC
        __syncthreads();                  // all waves done with buf before overwrite
        buf ^= 1;
#endif
    }
#pragma unroll
    for (int half = 0; half < 2; ++half) {
#pragma unroll
        for (int t = 0; t < 4; ++t) {
#pragma unroll
            for (int r = 0; r < 8; ++r) {
                const int row = mw + half * 16 + r + 8 * hi;
                const int col = nbase + t * 16 + lh;
                const float val = acc[half][t][r];
                if (headSplit) {
                    const int b = row >> 10, npos = row & (NN - 1);
                    const int h = col >> 6,  dh   = col & (DHH - 1);
                    C[(((size_t)(b * HH + h) * NN) + npos) * DHH + dh] = val;
                } else {
                    C[(size_t)row * Nc + col] = val;
                }
            }
        }
    }
}

// ---------------------------------------------------------------------------
// Flash attention (causal), per (b,h) x 64-query block, 4 waves x 16 rows.
// Q/K RoPE'd bf16 [B,H,N,DH]; Vt bf16 [B,H,DH,N]; O fp32 [B,H,N,DH].
// QK^T: A = Q (global), B = K^T (K row-major == B-frag order, global).
// PV:   A = P staged via per-wave LDS, B = Vt rows (global, key-contiguous).
// No block barriers needed: only wave-private LDS (in-order DScnt).
// ---------------------------------------------------------------------------
__global__ __launch_bounds__(128) void attn_kernel(
    const unsigned short* __restrict__ Q, const unsigned short* __restrict__ Kx,
    const unsigned short* __restrict__ Vt, float* __restrict__ O)
{
    __shared__ __align__(16) unsigned short Pw[4][16 * 32];  // per-wave P tile
    const int tid  = threadIdx.x;
    const int wave = tid >> 5, lane = tid & 31;
    const int lh = lane & 15, hi = lane >> 4;
    const int bh    = blockIdx.y;
    const int qbase = blockIdx.x * 64;
    const int qrow0 = qbase + wave * 16;
    const size_t base = (size_t)bh * NN * DHH;   // same footprint for Q/K/Vt/O
    const int ka0 = hi ? 8 : 0;
    const int kb0 = hi ? 8 : 0;

    // Q fragments (row = qrow0+lh, dh chunks 0-31 / 32-63), loaded once
    const unsigned short* qrow = Q + base + (size_t)(qrow0 + lh) * DHH;
    F16x16 qf[2];
#pragma unroll
    for (int c = 0; c < 2; ++c) {
        qf[c].q[0] = *(const uint4*)(qrow + c * 32 + ka0);
        qf[c].q[1] = *(const uint4*)(qrow + c * 32 + ka0 + 16);
    }

    v8f acc[4] = {};
    float mrow[8], lrow[8];
#pragma unroll
    for (int r = 0; r < 8; ++r) { mrow[r] = -3.0e38f; lrow[r] = 0.f; }

    const int iters = (qbase >> 5) + 2;       // causal: keys < qbase + 64
    for (int it = 0; it < iters; ++it) {
        const int jb = it * 32;
        // prefetch next key block rows into cache (global_prefetch_b8)
        if (it + 1 < iters)
            __builtin_prefetch((const void*)(Kx + base + (size_t)(jb + 32 + lane) * DHH), 0, 1);

        // S = (Q K^T) * scale for 2 key sub-tiles of 16
        v8f s[2];
#pragma unroll
        for (int st = 0; st < 2; ++st) {
            const unsigned short* krow = Kx + base + (size_t)(jb + st * 16 + lh) * DHH + kb0;
            F16x16 kf0, kf1;
            kf0.q[0] = *(const uint4*)(krow);       kf0.q[1] = *(const uint4*)(krow + 16);
            kf1.q[0] = *(const uint4*)(krow + 32);  kf1.q[1] = *(const uint4*)(krow + 48);
            v8f z = {};
            z = WMMA_BF16(qf[0].v, kf0.v, z);
            z = WMMA_BF16(qf[1].v, kf1.v, z);
            s[st] = z;
        }
        // causal mask + online softmax (row lives across 16 lanes of half-wave)
#pragma unroll
        for (int r = 0; r < 8; ++r) {
            const int row  = qrow0 + r + 8 * hi;
            const int rowl = r + 8 * hi;
            float s0 = s[0][r] * 0.125f;
            float s1 = s[1][r] * 0.125f;
            if (jb + lh > row)      s0 = -3.0e38f;
            if (jb + 16 + lh > row) s1 = -3.0e38f;
            float mx = fmaxf(s0, s1);
#pragma unroll
            for (int m = 1; m < 16; m <<= 1) mx = fmaxf(mx, __shfl_xor(mx, m, 32));
            const float mnew = fmaxf(mrow[r], mx);
            const float corr = __expf(mrow[r] - mnew);
            const float p0 = __expf(s0 - mnew);
            const float p1 = __expf(s1 - mnew);
            float rs = p0 + p1;
#pragma unroll
            for (int m = 1; m < 16; m <<= 1) rs += __shfl_xor(rs, m, 32);
            lrow[r] = lrow[r] * corr + rs;
            mrow[r] = mnew;
#pragma unroll
            for (int t = 0; t < 4; ++t) acc[t][r] *= corr;
            Pw[wave][rowl * 32 + lh]      = f2bf(p0);
            Pw[wave][rowl * 32 + 16 + lh] = f2bf(p1);
        }
        // O += P @ V   (wave-private LDS for P; Vt rows direct from global)
        F16x16 pf;
        const unsigned short* pp = &Pw[wave][lh * 32 + ka0];
        pf.q[0] = *(const uint4*)(pp);
        pf.q[1] = *(const uint4*)(pp + 16);
#pragma unroll
        for (int t = 0; t < 4; ++t) {
            F16x16 vf_;
            const unsigned short* vp = Vt + base + (size_t)(t * 16 + lh) * NN + jb + kb0;
            vf_.q[0] = *(const uint4*)(vp);
            vf_.q[1] = *(const uint4*)(vp + 16);
            acc[t] = WMMA_BF16(pf.v, vf_.v, acc[t]);
        }
    }
    // normalize + store
#pragma unroll
    for (int r = 0; r < 8; ++r) {
        const float inv = 1.0f / lrow[r];
        const int row = qrow0 + r + 8 * hi;
        float* orow = O + base + (size_t)row * DHH;
#pragma unroll
        for (int t = 0; t < 4; ++t) orow[t * 16 + lh] = acc[t][r] * inv;
    }
}

// ---------------------------------------------------------------------------
// host-side orchestration
// ---------------------------------------------------------------------------
extern "C" void kernel_launch(void* const* d_in, const int* in_sizes, int n_in,
                              void* d_out, int out_size, void* d_ws, size_t ws_size,
                              hipStream_t stream)
{
    (void)in_sizes; (void)n_in; (void)out_size; (void)ws_size;
    const float* tokens = (const float*)d_in[0];
    const float* gamma  = (const float*)d_in[1];
    const float* Wq     = (const float*)d_in[2];
    const float* Wk     = (const float*)d_in[3];
    const float* Wv     = (const float*)d_in[4];
    const float* Wo     = (const float*)d_in[5];
    float* outp = (float*)d_out;

    char* ws = (char*)d_ws;
    size_t off = 0;
    auto alloc = [&](size_t bytes) -> char* {
        char* p = ws + off;
        off += (bytes + 255) & ~(size_t)255;
        return p;
    };
    unsigned short* xbf  = (unsigned short*)alloc((size_t)NT * DD * 2);
    unsigned short* wqbf = (unsigned short*)alloc((size_t)DD * HD * 2);  // [Nc,K]
    unsigned short* wkbf = (unsigned short*)alloc((size_t)NLAYERS * DD * HD * 2);
    unsigned short* wvbf = (unsigned short*)alloc((size_t)NLAYERS * DD * HD * 2);
    unsigned short* wobf = (unsigned short*)alloc((size_t)HD * DD * 2);
    float* q0  = (float*)alloc((size_t)NT * HD * 4);
    float* q1  = (float*)alloc((size_t)NT * HD * 4);
    float* tmp = (float*)alloc((size_t)NT * HD * 4);
    unsigned short* qbf = (unsigned short*)alloc((size_t)NT * HD * 2);
    unsigned short* kbf = (unsigned short*)alloc((size_t)NT * HD * 2);
    unsigned short* vbf = (unsigned short*)alloc((size_t)NT * HD * 2);  // [B,H,DH,N]
    unsigned short* mrg = (unsigned short*)alloc((size_t)NT * HD * 2);

    // RMSNorm + one-shot weight transposes to bf16 [Nc,K]
    rmsnorm_kernel<<<NT, 256, 0, stream>>>(tokens, gamma, xbf);
    const int w1 = DD * HD;                    // 1M elements per matrix
    const dim3 tgrid(HD / 32, DD / 32);        // (32, 32)
    transpose_bf_kernel<<<tgrid, 256, 0, stream>>>(Wq, wqbf, DD, HD);
    for (int l = 0; l < NLAYERS; ++l) {
        transpose_bf_kernel<<<tgrid, 256, 0, stream>>>(Wk + (size_t)l * w1, wkbf + (size_t)l * w1, DD, HD);
        transpose_bf_kernel<<<tgrid, 256, 0, stream>>>(Wv + (size_t)l * w1, wvbf + (size_t)l * w1, DD, HD);
    }
    transpose_bf_kernel<<<tgrid, 256, 0, stream>>>(Wo, wobf, HD, DD);

    const dim3 ggrid(HD / 64, NT / 128);       // (16, 32)
    gemm_bf16_kernel<<<ggrid, 128, 0, stream>>>(xbf, wqbf, q0, NT, DD, HD, 1);

    const int pairTotal = BB * HH * NN * (DHH / 2);  // 2,097,152
    const int elemTotal = BB * HH * NN * DHH;        // 4,194,304
    float* qcur = q0; float* qnext = q1;
    for (int i = 0; i < NLAYERS; ++i) {
        gemm_bf16_kernel<<<ggrid, 128, 0, stream>>>(xbf, wkbf + (size_t)i * w1, tmp, NT, DD, HD, 1);
        rope_conv_kernel<<<pairTotal / 256, 256, 0, stream>>>(tmp, kbf, pairTotal);
        gemm_bf16_kernel<<<ggrid, 128, 0, stream>>>(xbf, wvbf + (size_t)i * w1, tmp, NT, DD, HD, 1);
        conv_vT_kernel<<<elemTotal / 256, 256, 0, stream>>>(tmp, vbf, elemTotal);
        rope_conv_kernel<<<pairTotal / 256, 256, 0, stream>>>(qcur, qbf, pairTotal);
        attn_kernel<<<dim3(NN / 64, BB * HH), 128, 0, stream>>>(qbf, kbf, vbf, qnext);
        if (i < NLAYERS - 1)
            silu_kernel<<<elemTotal / 256, 256, 0, stream>>>(qnext, elemTotal);
        float* t2 = qcur; qcur = qnext; qnext = t2;
    }
    merge_kernel<<<elemTotal / 256, 256, 0, stream>>>(qcur, mrg, elemTotal);
    gemm_bf16_kernel<<<ggrid, 128, 0, stream>>>(mrg, wobf, outp, NT, HD, DD, 0);
}
